// CrfRnn_12154757447824
// MI455X (gfx1250) — matrix operand order repaired and checked by reading, not compiled, not run on hardware
//
#include <hip/hip_runtime.h>

#define N_PIX 9216
#define IMG_W 96
#define L_LAB 21
#define C_PAD 32
#define CRF_EPS 1e-8f

// exp(-0.5*d2) == exp2(-(s*d)^2 summed), with s = sqrt(0.5*log2(e))
#define FSCALE 0.84932180028801904f
#define SG (FSCALE / 3.0f)     // grid coords / gamma,  pre-scaled
#define SA (FSCALE / 160.0f)   // grid coords / alpha,  pre-scaled
#define SB (FSCALE / 3.0f)     // rgb / beta, pre-scaled

typedef __attribute__((ext_vector_type(16))) _Float16 v16h;
typedef __attribute__((ext_vector_type(8)))  float    v8f;
typedef __attribute__((ext_vector_type(4)))  float    v4f;

// native base-2 exp: lowers to v_exp_f32
static __device__ __forceinline__ float fast_exp2(float x) {
    return __builtin_amdgcn_exp2f(x);
}

// ---------------------------------------------------------------------------
// Pre-scaled per-pixel features:
//   feat4[j] = { x*SG, y*SG, x*SA, y*SA }   (spatial for K_s and K_b)
//   rgb4[j]  = { r*SB, g*SB, b*SB, 0 }      (color for K_b)
// ---------------------------------------------------------------------------
__global__ void crf_feat_kernel(const float* __restrict__ img,
                                float* __restrict__ feat4, float* __restrict__ rgb4) {
    int j = blockIdx.x * blockDim.x + threadIdx.x;
    if (j >= N_PIX) return;
    int x = j % IMG_W, y = j / IMG_W;
    v4f f, c;
    f.x = (float)x * SG;
    f.y = (float)y * SG;
    f.z = (float)x * SA;
    f.w = (float)y * SA;
    c.x = img[0 * N_PIX + j] * SB;
    c.y = img[1 * N_PIX + j] * SB;
    c.z = img[2 * N_PIX + j] * SB;
    c.w = 0.0f;
    ((v4f*)feat4)[j] = f;
    ((v4f*)rgb4)[j]  = c;
}

// ---------------------------------------------------------------------------
// Softmax over 21 labels per pixel -> channel-major f16 q padded to 32
// channels.  Channel 21 = 1.0 (normalizer channel), 22..31 = 0.
// ---------------------------------------------------------------------------
__global__ void crf_softmax_kernel(const float* __restrict__ cur, _Float16* __restrict__ q16) {
    int j = blockIdx.x * blockDim.x + threadIdx.x;
    if (j >= N_PIX) return;
    float v[L_LAB];
    float m = -1e30f;
#pragma unroll
    for (int l = 0; l < L_LAB; ++l) { v[l] = cur[l * N_PIX + j]; m = fmaxf(m, v[l]); }
    float s = 0.0f;
#pragma unroll
    for (int l = 0; l < L_LAB; ++l) { v[l] = __expf(v[l] - m); s += v[l]; }
    float inv = 1.0f / s;
#pragma unroll
    for (int l = 0; l < L_LAB; ++l) q16[(size_t)l * N_PIX + j] = (_Float16)(v[l] * inv);
    q16[(size_t)L_LAB * N_PIX + j] = (_Float16)1.0f;
#pragma unroll
    for (int c = L_LAB + 1; c < C_PAD; ++c) q16[(size_t)c * N_PIX + j] = (_Float16)0.0f;
}

// ---------------------------------------------------------------------------
// Flash-style dense Gaussian filter.  Wave owns 16 output pixels (A: M rows).
// Since 96 % 32 == 0, every 32-pixel i-block sits in ONE image row: y_i is
// wave-uniform and x_i is affine in the compile-time element index, so the
// spatial part of d2 costs ~2 FMAs/entry and only RGB is loaded (1 b128/i).
// Kernel values go straight into the 16-bit 16x32 A layout; q^T rows are the
// 32x16 B tiles; 4x v_wmma_f32_16x16x32_f16 per i-block accumulate
// 21 labels + normalizer in f32.
// ---------------------------------------------------------------------------
__global__ __launch_bounds__(256) void crf_filter_kernel(
    const float* __restrict__ feat4, const float* __restrict__ rgb4,
    const _Float16* __restrict__ q16,
    float* __restrict__ filt_s, float* __restrict__ filt_b) {

    const int lane  = threadIdx.x & 31;
    const int wave  = threadIdx.x >> 5;
    const int jbase = blockIdx.x * 128 + wave * 16;
    const int jm    = lane & 15;      // A: M row owned by this lane; B: channel
    const int grp   = lane >> 4;      // lane group -> K sub-range
    const int j     = jbase + jm;

    const v4f fj = ((const v4f*)feat4)[j];
    const v4f cj = ((const v4f*)rgb4)[j];

    v8f accS0 = {}, accS1 = {}, accB0 = {}, accB1 = {};

    for (int ib = 0; ib < N_PIX; ib += 32) {
        // --- block-uniform spatial terms (one image row per 32-block) ------
        const int yrow = ib / IMG_W;            // scalar
        const int x0   = ib - yrow * IMG_W;     // scalar
        const float dyg = (float)yrow * SG - fj.y;
        const float dya = (float)yrow * SA - fj.w;
        const float ndyg2 = dyg * (-dyg);       // neg folded as src modifier
        const float ndya2 = dya * (-dya);
        const float xg0 = (float)(x0 + 16 * grp) * SG - fj.x;
        const float xa0 = (float)(x0 + 16 * grp) * SA - fj.z;

        v16h aS, aB;
#pragma unroll
        for (int e = 0; e < 16; ++e) {
            const int   kk = e;                             // compile-time
            const float ke = (float)kk;
            const int   i  = ib + kk + (grp << 4);

            const float dxg = fmaf(ke, SG, xg0);
            const float ts  = fmaf(dxg, -dxg, ndyg2);       // -(dx^2+dy^2) scaled
            aS[e] = (_Float16)fast_exp2(ts);                // v_exp_f32

            const v4f ci = ((const v4f*)rgb4)[i];
            const float dxa = fmaf(ke, SA, xa0);
            float tb = fmaf(dxa, -dxa, ndya2);
            const float dr = ci.x - cj.x, dg = ci.y - cj.y, db = ci.z - cj.z;
            tb = fmaf(dr, -dr, tb);
            tb = fmaf(dg, -dg, tb);
            tb = fmaf(db, -db, tb);
            aB[e] = (_Float16)fast_exp2(tb);
        }

        // --- B tiles (q^T, 32x16): lane = channel, 16 contiguous i's -------
        const _Float16* brow0 = q16 + (size_t)(jm)      * N_PIX + ib + grp * 16;
        const _Float16* brow1 = q16 + (size_t)(16 + jm) * N_PIX + ib + grp * 16;
        __builtin_prefetch(brow0 + 32, 0, 1);
        __builtin_prefetch(brow1 + 32, 0, 1);
        const v16h b0 = *(const v16h*)brow0;
        const v16h b1 = *(const v16h*)brow1;

        accS0 = __builtin_amdgcn_wmma_f32_16x16x32_f16(false, aS, false, b0, (short)0, accS0, false, false);
        accS1 = __builtin_amdgcn_wmma_f32_16x16x32_f16(false, aS, false, b1, (short)0, accS1, false, false);
        accB0 = __builtin_amdgcn_wmma_f32_16x16x32_f16(false, aB, false, b0, (short)0, accB0, false, false);
        accB1 = __builtin_amdgcn_wmma_f32_16x16x32_f16(false, aB, false, b1, (short)0, accB1, false, false);
    }

    // D layout: VGPR v -> M = v + 8*grp (pixel), lane&15 -> N (channel)
#pragma unroll
    for (int v = 0; v < 8; ++v) {
        const int jj = jbase + v + 8 * grp;
        filt_s[(size_t)(jm)      * N_PIX + jj] = accS0[v];
        filt_s[(size_t)(16 + jm) * N_PIX + jj] = accS1[v];
        filt_b[(size_t)(jm)      * N_PIX + jj] = accB0[v];
        filt_b[(size_t)(16 + jm) * N_PIX + jj] = accB1[v];
    }
}

// ---------------------------------------------------------------------------
// Normalize by the filtered ones-channel, apply the three 21x21 mixes
// (staged in LDS), add unary.
// ---------------------------------------------------------------------------
__global__ __launch_bounds__(256) void crf_epilogue_kernel(
    const float* __restrict__ filt_s, const float* __restrict__ filt_b,
    const float* __restrict__ unary,
    const float* __restrict__ Ws, const float* __restrict__ Wb, const float* __restrict__ Cm,
    float* __restrict__ cur_out) {

    __shared__ float sWs[L_LAB * L_LAB], sWb[L_LAB * L_LAB], sCm[L_LAB * L_LAB];
    for (int t = threadIdx.x; t < L_LAB * L_LAB; t += blockDim.x) {
        sWs[t] = Ws[t]; sWb[t] = Wb[t]; sCm[t] = Cm[t];
    }
    __syncthreads();

    int j = blockIdx.x * blockDim.x + threadIdx.x;
    if (j >= N_PIX) return;

    const float ns = 1.0f / (filt_s[(size_t)L_LAB * N_PIX + j] + CRF_EPS);
    const float nb = 1.0f / (filt_b[(size_t)L_LAB * N_PIX + j] + CRF_EPS);

    float ts[L_LAB], tb[L_LAB];
#pragma unroll
    for (int l = 0; l < L_LAB; ++l) {
        ts[l] = filt_s[(size_t)l * N_PIX + j] * ns;
        tb[l] = filt_b[(size_t)l * N_PIX + j] * nb;
    }
    float mix[L_LAB];
#pragma unroll 1
    for (int l = 0; l < L_LAB; ++l) {
        float a = 0.0f;
#pragma unroll
        for (int k = 0; k < L_LAB; ++k)
            a = fmaf(sWs[l * L_LAB + k], ts[k], fmaf(sWb[l * L_LAB + k], tb[k], a));
        mix[l] = a;
    }
#pragma unroll 1
    for (int l = 0; l < L_LAB; ++l) {
        float a = unary[(size_t)l * N_PIX + j];
#pragma unroll
        for (int m = 0; m < L_LAB; ++m) a = fmaf(sCm[l * L_LAB + m], mix[m], a);
        cur_out[(size_t)l * N_PIX + j] = a;
    }
}

// ---------------------------------------------------------------------------
extern "C" void kernel_launch(void* const* d_in, const int* in_sizes, int n_in,
                              void* d_out, int out_size, void* d_ws, size_t ws_size,
                              hipStream_t stream) {
    (void)in_sizes; (void)n_in; (void)out_size; (void)ws_size;
    const float* image  = (const float*)d_in[0];
    const float* logits = (const float*)d_in[1];
    const float* Ws     = (const float*)d_in[2];
    const float* Wb     = (const float*)d_in[3];
    const float* Cm     = (const float*)d_in[4];
    float* out = (float*)d_out;

    char* ws = (char*)d_ws;
    size_t off = 0;
    auto wsAlloc = [&](size_t bytes) -> void* {
        void* p = ws + off;
        off = (off + bytes + 255) & ~(size_t)255;
        return p;
    };
    float*    feat4  = (float*)   wsAlloc((size_t)N_PIX * 4 * sizeof(float));
    float*    rgb4   = (float*)   wsAlloc((size_t)N_PIX * 4 * sizeof(float));
    _Float16* q16    = (_Float16*)wsAlloc((size_t)C_PAD * N_PIX * sizeof(_Float16));
    float*    filt_s = (float*)   wsAlloc((size_t)C_PAD * N_PIX * sizeof(float));
    float*    filt_b = (float*)   wsAlloc((size_t)C_PAD * N_PIX * sizeof(float));
    float*    cur    = (float*)   wsAlloc((size_t)L_LAB * N_PIX * sizeof(float));

    crf_feat_kernel<<<(N_PIX + 255) / 256, 256, 0, stream>>>(image, feat4, rgb4);

    const float* src = logits;
    for (int it = 0; it < 5; ++it) {
        crf_softmax_kernel<<<(N_PIX + 255) / 256, 256, 0, stream>>>(src, q16);
        crf_filter_kernel<<<N_PIX / 128, 256, 0, stream>>>(feat4, rgb4, q16, filt_s, filt_b);
        float* dst = (it == 4) ? out : cur;
        crf_epilogue_kernel<<<(N_PIX + 255) / 256, 256, 0, stream>>>(
            filt_s, filt_b, logits, Ws, Wb, Cm, dst);
        src = dst;
    }
}
